// QuantizedLinear_14800457302576
// MI455X (gfx1250) — matrix operand order repaired and checked by reading, not compiled, not run on hardware
//
#include <hip/hip_runtime.h>

typedef __bf16 bf16;
typedef __attribute__((ext_vector_type(8)))  bf16  bf16x8;
typedef __attribute__((ext_vector_type(16))) bf16  bf16x16;
typedef __attribute__((ext_vector_type(8)))  float floatx8;

constexpr int GROUP = 64;

// ---------------- kernel 1: inverse permutation ----------------
__global__ void k_inv_perm(const int* __restrict__ q_perm, int* __restrict__ inv, int I) {
    int k = blockIdx.x * 256 + threadIdx.x;
    if (k < I) inv[q_perm[k]] = k;
}

// ---------------- kernel 2: xb[m,j] = bf16(x[m,j] * cs[j]) ----------------
__global__ void k_prep_x(const float* __restrict__ x, const float* __restrict__ cs,
                         bf16* __restrict__ xb, int M, int K) {
    int perRow = K >> 3;
    long long t = (long long)blockIdx.x * 256 + threadIdx.x; // one thread per 8 elems
    if (t >= (long long)M * perRow) return;
    int m  = (int)(t / perRow);
    int j0 = (int)(t % perRow) << 3;
    const float4* xp = (const float4*)(x + (size_t)m * K + j0);
    const float4* cp = (const float4*)(cs + j0);
    float4 a0 = xp[0], a1 = xp[1];
    float4 c0 = cp[0], c1 = cp[1];
    float v[8] = {a0.x*c0.x, a0.y*c0.y, a0.z*c0.z, a0.w*c0.w,
                  a1.x*c1.x, a1.y*c1.y, a1.z*c1.z, a1.w*c1.w};
    union { bf16x8 v8; bf16 e[8]; } r;
#pragma unroll
    for (int u = 0; u < 8; ++u) r.e[u] = (bf16)v[u];
    *(bf16x8*)(xb + (size_t)m * K + j0) = r.v8;
}

// -------- kernel 3: wb[o,j] = bf16(dequant(qweight[o, inv[j]])) (perm folded) -----
__global__ void k_prep_w(const unsigned* __restrict__ qw, const float* __restrict__ scales,
                         const float* __restrict__ zeros, const int* __restrict__ inv,
                         bf16* __restrict__ wb, int O, int K) {
    int perRow = K >> 3;
    long long t = (long long)blockIdx.x * 256 + threadIdx.x;
    if (t >= (long long)O * perRow) return;
    int o  = (int)(t / perRow);
    int j0 = (int)(t % perRow) << 3;
    const unsigned* qrow = qw     + (size_t)o * (K >> 3);
    const float*    srow = scales + (size_t)o * (K / GROUP);
    const float*    zrow = zeros  + (size_t)o * (K / GROUP);
    union { bf16x8 v8; bf16 e[8]; } r;
#pragma unroll
    for (int u = 0; u < 8; ++u) {
        int k = inv[j0 + u];                       // gather within cached row
        unsigned w = qrow[k >> 3];
        float nib = (float)((w >> ((k & 7) * 4)) & 0xFu);
        int g = k / GROUP;
        r.e[u] = (bf16)(nib * srow[g] + zrow[g]);
    }
    *(bf16x8*)(wb + (size_t)o * K + j0) = r.v8;
}

// ---- CDNA5 async copy, GVS form: MEM[saddr + voff + ioff] -> LDS[vdst + ioff] ----
// Per-stage advance lives in the SGPR base: no VALU writes near WMMA operands.
__device__ __forceinline__ void async_copy32_s(unsigned lds, unsigned voff, const void* sbase) {
    asm volatile("global_load_async_to_lds_b128 %0, %1, %2\n\t"
                 "global_load_async_to_lds_b128 %0, %1, %2 offset:16"
                 :: "v"(lds), "v"(voff), "s"(sbase) : "memory");
}
__device__ __forceinline__ void wait_async0() {
    asm volatile("s_wait_asynccnt 0x0" ::: "memory");
}

// ---------------- kernel 4: C = A(MxK,bf16) * B(NxK,bf16)^T + bias ----------------
#define BM 128
#define BN 128
#define BK 32
#define LDAS 40   // padded LDS row stride (bf16) -> conflict-free bank spread

__global__ __launch_bounds__(256)
void k_gemm(const bf16* __restrict__ A, const bf16* __restrict__ B,
            const float* __restrict__ bias, float* __restrict__ C,
            int M, int N, int K) {
    __shared__ bf16 As[2][BM * LDAS];   // 2*128*40*2 = 20480 B
    __shared__ bf16 Bs[2][BN * LDAS];   // 20480 B

    const int tid   = threadIdx.x;
    const int lane  = tid & 31;
    const int wave  = tid >> 5;
    const int wm    = wave & 3;          // 4 waves along M (32 rows each)
    const int wn    = wave >> 2;         // 2 waves along N (64 cols each)
    const int lmod  = lane & 15;
    const int lhalf = lane >> 4;
    const int m0 = blockIdx.y * BM;
    const int n0 = blockIdx.x * BN;

    // global->LDS mapping: 2 threads per tile row, 16 bf16 (32B) each
    const int grow = tid >> 1;           // 0..127
    const int gcol = (tid & 1) << 4;     // 0 or 16
    // per-lane 32-bit byte offsets (constant); per-stage advance goes in SGPR base
    const unsigned voffA = (unsigned)(((size_t)(m0 + grow) * K + gcol) * sizeof(bf16));
    const unsigned voffB = (unsigned)(((size_t)(n0 + grow) * K + gcol) * sizeof(bf16));
    const int lds_off = grow * LDAS + gcol;

    // LDS byte addresses (low 32 bits of generic pointer = LDS offset)
    const unsigned ldsA0 = (unsigned)(size_t)&As[0][lds_off];
    const unsigned ldsB0 = (unsigned)(size_t)&Bs[0][lds_off];
    const unsigned bufStride = (unsigned)(BM * LDAS * sizeof(bf16)); // 10240

    floatx8 acc[2][4] = {};

    auto compute = [&](int buf) {
        union F { bf16x16 v; bf16x8 h[2]; };
        F a[2], b[4];
#pragma unroll
        for (int i = 0; i < 2; ++i) {
            const bf16* p = &As[buf][(wm * 32 + i * 16 + lmod) * LDAS];
            a[i].h[0] = *(const bf16x8*)(p + lhalf * 8);        // K = half*8 + 0..7
            a[i].h[1] = *(const bf16x8*)(p + 16 + lhalf * 8);   // K = 16 + half*8 + 0..7
        }
#pragma unroll
        for (int j = 0; j < 4; ++j) {
            const bf16* p = &Bs[buf][(wn * 64 + j * 16 + lmod) * LDAS + lhalf * 8];
            b[j].h[0] = *(const bf16x8*)(p);                    // K = half*8 + 0..7
            b[j].h[1] = *(const bf16x8*)(p + 16);               // K = 16 + half*8 + 0..7
        }
        // j-major so both A fragments are consumed immediately
#pragma unroll
        for (int j = 0; j < 4; ++j)
#pragma unroll
            for (int i = 0; i < 2; ++i)
                acc[i][j] = __builtin_amdgcn_wmma_f32_16x16x32_bf16(
                    false, a[i].v, false, b[j].v, (short)0, acc[i][j], false, false);
    };

    // prologue: stage 0 via async DMA to LDS
    const char* baseA = (const char*)A;
    const char* baseB = (const char*)B;
    async_copy32_s(ldsA0, voffA, baseA);
    async_copy32_s(ldsB0, voffB, baseB);
    wait_async0();
    __syncthreads();

    const int NT = K / BK;
    // steady state: always prefetch stage kt+1 (last iteration peeled)
    for (int kt = 0; kt < NT - 1; ++kt) {
        const unsigned sel = ((kt + 1) & 1) ? bufStride : 0u;
        const size_t stageOff = (size_t)(kt + 1) * (BK * sizeof(bf16)); // +64 B/stage
        async_copy32_s(ldsA0 + sel, voffA, baseA + stageOff);
        async_copy32_s(ldsB0 + sel, voffB, baseB + stageOff);

        compute(kt & 1);

        wait_async0();                  // stage kt+1 landed in LDS
        __syncthreads();
    }
    compute((NT - 1) & 1);              // final stage: nothing left to prefetch

    // epilogue: C/D layout -> row = v + 8*lhalf (+tile), col = lmod (+tile)
#pragma unroll
    for (int j = 0; j < 4; ++j) {
        const int col = n0 + wn * 64 + j * 16 + lmod;
        const float bj = bias[col];
#pragma unroll
        for (int i = 0; i < 2; ++i) {
            const int rbase = m0 + wm * 32 + i * 16 + lhalf * 8;
#pragma unroll
            for (int v = 0; v < 8; ++v)
                C[(size_t)(rbase + v) * N + col] = acc[i][j][v] + bj;
        }
    }
}

extern "C" void kernel_launch(void* const* d_in, const int* in_sizes, int n_in,
                              void* d_out, int out_size, void* d_ws, size_t ws_size,
                              hipStream_t stream) {
    const float*    x      = (const float*)d_in[0];
    const float*    cs     = (const float*)d_in[1];
    const unsigned* qw     = (const unsigned*)d_in[2];
    const int*      qperm  = (const int*)d_in[3];
    const float*    scales = (const float*)d_in[4];
    const float*    zeros  = (const float*)d_in[5];
    const float*    bias   = (const float*)d_in[6];
    float*          out    = (float*)d_out;

    const int I = in_sizes[1];            // channel_scale: (1,1,I)
    const int O = in_sizes[6];            // bias: (O,)
    const int M = in_sizes[0] / I;        // B*S
    const int K = I, N = O;

    // workspace carve-up: inv_perm | xb (M*K bf16) | wb (O*K bf16)
    char* ws  = (char*)d_ws;
    int*  inv = (int*)ws;
    size_t off = (((size_t)I * sizeof(int)) + 255) & ~(size_t)255;
    bf16* xb = (bf16*)(ws + off);
    bf16* wb = xb + (size_t)M * K;

    k_inv_perm<<<(I + 255) / 256, 256, 0, stream>>>(qperm, inv, I);

    long long nx = (long long)M * (K / 8);
    k_prep_x<<<(unsigned)((nx + 255) / 256), 256, 0, stream>>>(x, cs, xb, M, K);

    long long nw = (long long)O * (K / 8);
    k_prep_w<<<(unsigned)((nw + 255) / 256), 256, 0, stream>>>(qw, scales, zeros, inv, wb, O, K);

    dim3 grid(N / BN, M / BM);
    k_gemm<<<grid, 256, 0, stream>>>(xb, wb, bias, out, M, N, K);
}
